// Combined_LoTeConv_12979391169075
// MI455X (gfx1250) — matrix operand order repaired and checked by reading, not compiled, not run on hardware
//
#include <hip/hip_runtime.h>
#include <hip/hip_bf16.h>

// ---------------------------------------------------------------------------
// MPS network on MI455X (gfx1250).
//
// Roofline: the mats einsum streams ~295MB of f32 cores exactly once;
// 16 FLOP/byte intensity => ~373 TFLOP/s needed at 23.3 TB/s, so we use
// v_wmma_f32_16x16x32_f16 and make the kernel purely HBM-bound.
// Cores are streamed with the Tensor Data Mover (tensor_load_to_lds,
// TENSORcnt), double buffered so compute on chunk q overlaps DMA of q+1.
// The per-patch label block is staged with global_load_async_to_lds_b128
// (ASYNCcnt), overlapping the phi gather.
// ---------------------------------------------------------------------------

typedef __attribute__((ext_vector_type(16))) _Float16     v16h;
typedef __attribute__((ext_vector_type(8)))  float        v8f;
typedef __attribute__((ext_vector_type(4)))  unsigned int u32x4;
typedef __attribute__((ext_vector_type(8)))  int          i32x8;
typedef __attribute__((ext_vector_type(4)))  int          i32x4;

#define BATCH 32

// LDS byte offset of a shared-memory pointer: low 32 bits of the flat
// shared-aperture address (ISA 10.2).
__device__ __forceinline__ unsigned lds_off(const void* p)
{
    return (unsigned)(unsigned long long)p;
}

__device__ __forceinline__ void wait_async_all()
{
    asm volatile("s_wait_asynccnt 0x0" ::: "memory");
}

// TDM: DMA one cores K-chunk (tile 32 rows x 256 cols f32) into LDS.
// tensor_dim1 = rows (= F - kc*32): rows beyond the tensor are zero-filled
// by the TDM (ISA 8.2 OOB reads return zero), which handles F % 32 != 0
// tails with no masking in the compute loop.  D# per ISA 8.3/8.4.
__device__ __forceinline__ void tdm_load_chunk(const float* gaddr, unsigned ldsAddr,
                                               int rows)
{
    unsigned long long ga = (unsigned long long)gaddr;
    u32x4 g0;
    g0[0] = 1u;                                            // count=1 (user D#)
    g0[1] = ldsAddr;                                       // lds_addr [63:32]
    g0[2] = (unsigned)(ga & 0xffffffffu);                  // global_addr lo
    g0[3] = (unsigned)((ga >> 32) & 0x1ffffffu)            // global_addr [56:32]
          | (2u << 30);                                    // type = 2 ("image")
    i32x8 g1;
    g1[0] = (int)(2u << 16);                               // data_size = 4B
    g1[1] = (int)(256u << 16);                             // tensor_dim0 = 256
    g1[2] = (int)(((unsigned)rows & 0xffffu) << 16);       // tensor_dim1 = rows
    g1[3] = (int)(256u << 16);                             // tile_dim0 = 256
    g1[4] = 32;                                            // tile_dim1 = 32
    g1[5] = 256;                                           // tensor_dim0_stride
    g1[6] = 0;
    g1[7] = 0;
    i32x4 z4 = {0, 0, 0, 0};                               // 2-D tile: groups 2/3 unused
    i32x8 z8 = {0, 0, 0, 0, 0, 0, 0, 0};
    __builtin_amdgcn_tensor_load_to_lds(g0, g1, z4, z4, z8, 0);
}

// Store one phi value into the pre-fragmented A layout:
// fragment element e of lane holds K = kc*32 + (e&7) + ((e>>3)<<4) + ((lane>>4)<<3),
// row M = (mt*16 + lane&15).  Inverse map from (k, m):
__device__ __forceinline__ void store_phi(_Float16* phi2, int KC, int l, int k, int m,
                                          _Float16 v)
{
    int kc   = k >> 5, kl = k & 31;
    int e    = (kl & 7) | ((kl >> 4) << 3);
    int half = (kl >> 3) & 1;
    int lane = (m & 15) | (half << 4);
    int mt   = m >> 4;
    phi2[((((size_t)l * KC + kc) * 2 + mt) * 32 + lane) * 16 + e] = v;
}

// ---------------------------------------------------------------------------
// Generic MPS kernel:  xin [B, C, P, L] f32, cores [P, L, F, 16, 16] f32,
// label [P, 16, O, 16] f32  ->  out [B, P, O] f32.   One workgroup per patch,
// 256 threads = 8 waves; wave w owns output columns [32w, 32w+32).
// ---------------------------------------------------------------------------
__global__ void __launch_bounds__(256)
mps_wmma_kernel(const float* __restrict__ xin,
                const float* __restrict__ cores,
                const float* __restrict__ label,
                float* __restrict__ out,
                int C, int F, int Fpad, int P, int L, int O)
{
    extern __shared__ char smem[];
    float*    mats   = (float*)smem;                    // 256*33 f32 (bank pad)
    float*    labelR = mats + 256 * 33;                 // 16*16 f32
    float*    cbuf0  = labelR + 256;                    // 8192 f32 stream buf A
    float*    cbuf1  = cbuf0 + 8192;                    // 8192 f32 stream buf B
    _Float16* phi2   = (_Float16*)(cbuf1 + 8192);       // L*Fpad*32 f16 (frag layout)

    const int p    = blockIdx.x;
    const int tid  = threadIdx.x;
    const int lane = tid & 31;
    const int half = lane >> 4;
    const int nn   = lane & 15;
    const int nb0  = (tid >> 5) * 32;
    const int KC   = Fpad >> 5;
    const bool firstWave = (__builtin_amdgcn_readfirstlane(tid) < 32);

    const float* coresP = cores + (size_t)p * L * (size_t)F * 256;

    // ---- kick off TDM of the first cores chunk (wave 0 only, scalar guard:
    //      tensor ops ignore EXEC, so every wave reaching this would issue) ----
    if (firstWave)
        tdm_load_chunk(coresP, lds_off(cbuf0), F);

    // ---- async-stage this patch's label block into cbuf1 (ASYNCcnt) ----
    {
        unsigned lb = lds_off(cbuf1);
        const float* lblk = label + (size_t)p * 16 * O * 16;
        int nf4 = 16 * O * 4;                         // # of 16B pieces
        for (int t4 = tid; t4 < nf4; t4 += 256) {
            unsigned long long ga = (unsigned long long)(lblk + t4 * 4);
            unsigned lo = lb + (unsigned)(t4 << 4);
            asm volatile("global_load_async_to_lds_b128 %0, %1, off"
                         :: "v"(lo), "v"(ga) : "memory");
        }
    }

    // ---- stage phi = [x, 1-x] into LDS in A-fragment layout, f16 ----
    {
        const int CL = C * L;
        for (int t = tid; t < BATCH * CL; t += 256) {
            int l = t % L;
            int c = (t / L) % C;
            int m = t / CL;                     // batch index
            float xv = xin[(((size_t)m * C + c) * P + p) * L + l];
            store_phi(phi2, KC, l, c,     m, (_Float16)xv);
            store_phi(phi2, KC, l, C + c, m, (_Float16)(1.0f - xv));
        }
        int pad = Fpad - 2 * C;
        if (pad > 0) {
            for (int t = tid; t < L * pad * 32; t += 256) {
                int m   = t & 31;
                int rem = t >> 5;
                int k   = 2 * C + rem % pad;
                int l   = rem / pad;
                store_phi(phi2, KC, l, k, m, (_Float16)0.0f);
            }
        }
    }

    wait_async_all();
    __syncthreads();

    // ---- precontract label with right boundary RB (= 1/4 each) from LDS ----
    for (int t = tid; t < 16 * O; t += 256) {
        const float* lp = cbuf1 + t * 16;
        float s = 0.0f;
#pragma unroll
        for (int j = 0; j < 16; ++j) s += lp[j];
        labelR[(t / O) * 16 + (t % O)] = 0.25f * s;
    }

    __builtin_amdgcn_s_wait_tensorcnt(0);
    __syncthreads();

    // ---- left boundary vector LB = 1/4 (wave-0 registers, lane = batch) ----
    float v[16];
#pragma unroll
    for (int i = 0; i < 16; ++i) v[i] = 0.25f;

    int parity = 0;

    for (int l = 0; l < L; ++l) {
        v8f acc00{}, acc01{}, acc10{}, acc11{};     // [mtile][ntile]

        for (int kc = 0; kc < KC; ++kc) {
            // issue TDM for the next chunk into the other buffer (wave 0 only)
            {
                int kc2 = kc + 1, l2 = l;
                if (kc2 == KC) { kc2 = 0; ++l2; }
                if (l2 < L && firstWave) {
                    const float* gsrc = coresP + ((size_t)l2 * F + (size_t)kc2 * 32) * 256;
                    unsigned nlds = lds_off(parity ? cbuf0 : cbuf1);
                    tdm_load_chunk(gsrc, nlds, F - kc2 * 32);
                }
            }

            // A fragments: 2 x 32B contiguous per lane (ds_load_b128 pairs)
            const _Float16* pfr = phi2 + ((size_t)(l * KC + kc) * 2) * 512;
            v16h a0 = *(const v16h*)(pfr + lane * 16);
            v16h a1 = *(const v16h*)(pfr + 512 + lane * 16);

            // B fragments from the staged LDS chunk (f32 -> f16); rows beyond F
            // were zero-filled by the TDM, so no masking is needed.
            const float* cb = parity ? cbuf1 : cbuf0;
            int colB = nb0 + nn;
            v16h b0, b1;
#pragma unroll
            for (int e = 0; e < 16; ++e) {
                int kl = (half << 3) + (e & 7) + ((e >> 3) << 4);
                b0[e] = (_Float16)cb[kl * 256 + colB];
                b1[e] = (_Float16)cb[kl * 256 + colB + 16];
            }

            acc00 = __builtin_amdgcn_wmma_f32_16x16x32_f16(false, a0, false, b0, (short)0, acc00, false, false);
            acc01 = __builtin_amdgcn_wmma_f32_16x16x32_f16(false, a0, false, b1, (short)0, acc01, false, false);
            acc10 = __builtin_amdgcn_wmma_f32_16x16x32_f16(false, a1, false, b0, (short)0, acc10, false, false);
            acc11 = __builtin_amdgcn_wmma_f32_16x16x32_f16(false, a1, false, b1, (short)0, acc11, false, false);

            // next chunk's DMA must land; everyone must be done with buf[parity]
            __builtin_amdgcn_s_wait_tensorcnt(0);
            __syncthreads();
            parity ^= 1;
        }

        // ---- scatter C tiles to LDS: mats[n*33 + m] ----
        // C layout: VGPR r -> row (r + half*8), col nn (within tile).
#pragma unroll
        for (int r = 0; r < 8; ++r) {
            int m0 = r + (half << 3);
            mats[(nb0 + nn)      * 33 + m0]      = acc00[r];
            mats[(nb0 + 16 + nn) * 33 + m0]      = acc01[r];
            mats[(nb0 + nn)      * 33 + 16 + m0] = acc10[r];
            mats[(nb0 + 16 + nn) * 33 + 16 + m0] = acc11[r];
        }
        __syncthreads();

        // ---- scan step (wave 0, lane = batch b):  v' = v * mats[b] ----
        if (tid < 32) {
            float vn[16];
#pragma unroll
            for (int j = 0; j < 16; ++j) vn[j] = 0.0f;
#pragma unroll
            for (int i = 0; i < 16; ++i) {
                float vi = v[i];
#pragma unroll
                for (int j = 0; j < 16; ++j)
                    vn[j] = fmaf(vi, mats[(i * 16 + j) * 33 + tid], vn[j]);
            }
#pragma unroll
            for (int j = 0; j < 16; ++j) v[j] = vn[j];
        }
        __syncthreads();
    }

    // ---- label contraction: out[b,p,o] = sum_i v[i] * labelR[i,o] ----
    if (tid < 32) {
        for (int o = 0; o < O; ++o) {
            float s = 0.0f;
#pragma unroll
            for (int i = 0; i < 16; ++i) s = fmaf(v[i], labelR[i * 16 + o], s);
            out[((size_t)tid * P + p) * O + o] = s;
        }
    }
}

// ---------------------------------------------------------------------------
// BatchNorm1d (training-mode batch stats) on [B, CH, L0+L1]; in1 may be null.
// ---------------------------------------------------------------------------
__global__ void __launch_bounds__(256)
bn_kernel(const float* __restrict__ in0, const float* __restrict__ in1,
          const float* __restrict__ g, const float* __restrict__ bb,
          float* __restrict__ out, int CH, int L0, int L1)
{
    __shared__ float buf[1024];
    __shared__ float r1[256];
    __shared__ float r2[256];
    const int ch  = blockIdx.x;
    const int tid = threadIdx.x;
    const int L   = L0 + L1;
    const int N   = BATCH * L;

    float s = 0.0f, s2 = 0.0f;
    for (int idx = tid; idx < N; idx += 256) {
        int b = idx / L, t = idx % L;
        float x = (t < L0) ? in0[((size_t)b * CH + ch) * L0 + t]
                           : in1[((size_t)b * CH + ch) * L1 + (t - L0)];
        buf[idx] = x;
        s += x; s2 += x * x;
    }
    r1[tid] = s; r2[tid] = s2;
    __syncthreads();
    for (int off = 128; off > 0; off >>= 1) {
        if (tid < off) { r1[tid] += r1[tid + off]; r2[tid] += r2[tid + off]; }
        __syncthreads();
    }
    float mean  = r1[0] / (float)N;
    float var   = r2[0] / (float)N - mean * mean;
    float scale = rsqrtf(var + 1e-5f) * g[ch];
    float shift = bb[ch];
    for (int idx = tid; idx < N; idx += 256) {
        int b = idx / L, t = idx % L;
        out[((size_t)b * CH + ch) * L + t] = (buf[idx] - mean) * scale + shift;
    }
}

// ---------------------------------------------------------------------------
// Packing / reshuffle kernels (unfold2 + reshape + transpose index maps).
// ---------------------------------------------------------------------------
__global__ void pack_lote1(const float* __restrict__ x, float* __restrict__ lx)
{
    int t = blockIdx.x * 256 + threadIdx.x;
    if (t >= BATCH * 49152) return;
    int b = t / 49152, f = t % 49152;
    int s = f & 15, r = (f >> 4) & 15, w = (f >> 8) & 7, h = (f >> 11) & 7, c = f >> 14;
    lx[t] = x[(((size_t)b * 3 + c) * 128 + h * 16 + r) * 128 + w * 16 + s];
}

__global__ void pack_conv1(const float* __restrict__ x, float* __restrict__ cx)
{
    int t = blockIdx.x * 256 + threadIdx.x;
    if (t >= BATCH * 49152) return;
    int b = t / 49152, f = t % 49152;
    int s = f & 3, r = (f >> 2) & 3, w = (f >> 4) & 31, h = (f >> 9) & 31, c = f >> 14;
    cx[t] = x[(((size_t)b * 3 + c) * 128 + h * 4 + r) * 128 + w * 4 + s];
}

__global__ void pack_s2(const float* __restrict__ lybn, const float* __restrict__ cybn,
                        float* __restrict__ comb, float* __restrict__ combT)
{
    int t = blockIdx.x * 256 + threadIdx.x;
    if (t >= BATCH * 16 * 64 * 20) return;
    int b = t / 20480, rem = t % 20480;
    int ch = rem / 1280, rem2 = rem % 1280;
    int p2 = rem2 / 20, l2 = rem2 % 20;
    float val;
    if (l2 < 4) {
        int fl = p2 * 4 + l2;
        int s8 = fl & 7, r8 = (fl >> 3) & 7, w2 = (fl >> 6) & 1, h2 = (fl >> 7) & 1;
        int u = h2 * 8 + r8, vv = w2 * 8 + s8;
        val = lybn[((size_t)b * 256 + ch * 16 + u) * 16 + vv];
    } else {
        int q = l2 - 4;
        int fl = p2 * 16 + ch;
        int s4 = fl & 3, r4 = (fl >> 2) & 3, w8 = (fl >> 4) & 7, h8 = (fl >> 7) & 7;
        int u = h8 * 4 + r4, vv = w8 * 4 + s4;
        int p1 = q * 64 + u * 2 + (vv >> 4), o = vv & 15;
        val = cybn[((size_t)b * 1024 + p1) * 16 + o];
    }
    comb[t] = val;
    combT[(((size_t)b * 20 + l2) * 64 + p2) * 16 + ch] = val;
}

__global__ void pack_s3_l(const float* __restrict__ bn2, float* __restrict__ lx3)
{
    int t = blockIdx.x * 256 + threadIdx.x;
    if (t >= BATCH * 32 * 16 * 4) return;
    int b = t / 2048, ch = (t >> 6) & 31, p16 = (t >> 2) & 15, l4 = t & 3;
    int fl = p16 * 4 + l4;
    int s = fl & 3, r = (fl >> 2) & 3, w2 = (fl >> 4) & 1, h2 = (fl >> 5) & 1;
    int u = h2 * 4 + r, vv = w2 * 4 + s;
    int idx = ch * 64 + u * 8 + vv;
    lx3[t] = bn2[((size_t)b * 64 + (idx >> 5)) * 32 + (idx & 31)];
}

__global__ void pack_s3_c(const float* __restrict__ bn2, float* __restrict__ cx3)
{
    int t = blockIdx.x * 256 + threadIdx.x;
    if (t >= BATCH * 2048) return;
    int b = t / 2048, fl = t % 2048;
    int s = fl & 1, r = (fl >> 1) & 1, w = (fl >> 2) & 3, h = (fl >> 4) & 3, c = (fl >> 6) & 31;
    int u = h * 2 + r, vv = w * 2 + s;
    int idx = c * 64 + u * 8 + vv;
    cx3[t] = bn2[((size_t)b * 64 + (idx >> 5)) * 32 + (idx & 31)];
}

__global__ void pack_final(const float* __restrict__ y3bn, float* __restrict__ xf)
{
    int t = blockIdx.x * 256 + threadIdx.x;
    if (t >= BATCH * 32 * 16) return;
    int b = t / 512, c = (t >> 4) & 31, l = t & 15;
    xf[t] = y3bn[((size_t)b * 16 + l) * 32 + c];
}

// ---------------------------------------------------------------------------
static inline void launch_mps(const float* xin, const float* cores, const float* label,
                              float* out, int C, int F, int Fpad, int P, int L, int O,
                              hipStream_t stream)
{
    size_t smem = (size_t)(256 * 33 + 256 + 2 * 8192) * sizeof(float)
                + (size_t)L * Fpad * 32 * sizeof(_Float16);
    mps_wmma_kernel<<<P, 256, smem, stream>>>(xin, cores, label, out, C, F, Fpad, P, L, O);
}

extern "C" void kernel_launch(void* const* d_in, const int* in_sizes, int n_in,
                              void* d_out, int out_size, void* d_ws, size_t ws_size,
                              hipStream_t stream)
{
    (void)in_sizes; (void)n_in; (void)out_size; (void)ws_size;

    const float* x   = (const float*)d_in[0];
    const float* l1c = (const float*)d_in[1];
    const float* l1l = (const float*)d_in[2];
    const float* c1c = (const float*)d_in[3];
    const float* c1l = (const float*)d_in[4];
    const float* l2c = (const float*)d_in[5];
    const float* l2l = (const float*)d_in[6];
    const float* c2c = (const float*)d_in[7];
    const float* c2l = (const float*)d_in[8];
    const float* l3c = (const float*)d_in[9];
    const float* l3l = (const float*)d_in[10];
    const float* c3c = (const float*)d_in[11];
    const float* c3l = (const float*)d_in[12];
    const float* fc  = (const float*)d_in[13];
    const float* fl  = (const float*)d_in[14];
    const float* g1  = (const float*)d_in[15];
    const float* b1  = (const float*)d_in[16];
    const float* gc1 = (const float*)d_in[17];
    const float* bc1 = (const float*)d_in[18];
    const float* g2  = (const float*)d_in[19];
    const float* b2  = (const float*)d_in[20];
    const float* g3  = (const float*)d_in[21];
    const float* b3  = (const float*)d_in[22];

    float* ws    = (float*)d_ws;
    float* lx    = ws;  ws += 1572864;   // [32,48,256,4]
    float* cx    = ws;  ws += 1572864;   // [32,16,1024,3]
    float* m1l   = ws;  ws += 131072;    // [32,256,16]
    float* m1c   = ws;  ws += 524288;    // [32,1024,16]
    float* lybn  = ws;  ws += 131072;
    float* cybn  = ws;  ws += 524288;
    float* comb  = ws;  ws += 655360;    // [32,16,64,20]
    float* combT = ws;  ws += 655360;    // [32,20,64,16]
    float* lo2   = ws;  ws += 32768;     // [32,64,16]
    float* co2   = ws;  ws += 32768;
    float* bn2o  = ws;  ws += 65536;     // [32,64,32]
    float* lx3   = ws;  ws += 65536;     // [32,32,16,4]
    float* cx3   = ws;  ws += 65536;     // [32,4,16,32]
    float* lo3   = ws;  ws += 8192;      // [32,16,16]
    float* co3   = ws;  ws += 8192;
    float* y3bn  = ws;  ws += 16384;     // [32,16,32]
    float* xf    = ws;  ws += 16384;     // [32,32,1,16]

    // ---- stage 1 ----
    pack_lote1<<<(BATCH * 49152 + 255) / 256, 256, 0, stream>>>(x, lx);
    pack_conv1<<<(BATCH * 49152 + 255) / 256, 256, 0, stream>>>(x, cx);
    launch_mps(lx, l1c, l1l, m1l, 48, 96, 96, 256, 4, 16, stream);
    launch_mps(cx, c1c, c1l, m1c, 16, 32, 32, 1024, 3, 16, stream);
    bn_kernel<<<256, 256, 0, stream>>>(m1l, nullptr, g1, b1, lybn, 256, 16, 0);
    bn_kernel<<<1024, 256, 0, stream>>>(m1c, nullptr, gc1, bc1, cybn, 1024, 16, 0);

    // ---- stage 2 ----
    pack_s2<<<(BATCH * 20480 + 255) / 256, 256, 0, stream>>>(lybn, cybn, comb, combT);
    launch_mps(comb,  l2c, l2l, lo2, 16, 32, 32, 64, 20, 16, stream);
    launch_mps(combT, c2c, c2l, co2, 20, 40, 64, 64, 16, 16, stream);
    bn_kernel<<<64, 256, 0, stream>>>(lo2, co2, g2, b2, bn2o, 64, 16, 16);

    // ---- stage 3 ----
    pack_s3_l<<<(BATCH * 2048 + 255) / 256, 256, 0, stream>>>(bn2o, lx3);
    pack_s3_c<<<(BATCH * 2048 + 255) / 256, 256, 0, stream>>>(bn2o, cx3);
    launch_mps(lx3, l3c, l3l, lo3, 32, 64, 64, 16, 4, 16, stream);
    launch_mps(cx3, c3c, c3l, co3, 4, 8, 32, 16, 32, 16, stream);
    bn_kernel<<<16, 256, 0, stream>>>(lo3, co3, g3, b3, y3bn, 16, 16, 16);

    // ---- final ----
    pack_final<<<(BATCH * 512 + 255) / 256, 256, 0, stream>>>(y3bn, xf);
    launch_mps(xf, fc, fl, (float*)d_out, 32, 64, 64, 1, 16, 10, stream);
}
